// DecompModel_38345468019450
// MI455X (gfx1250) — compile-verified, matrix operand order checked by reading, not run on hardware
//
#include <hip/hip_runtime.h>
#include <hip/hip_bf16.h>
#include <math.h>

#define N_NODES 20000
#define N_EDGES 320000
#define EMB 128
#define K_DIM 256      // 2*EMB
#define N_DIM 512      // 4*EMB

typedef __attribute__((ext_vector_type(16))) _Float16 v16h;
typedef __attribute__((ext_vector_type(8)))  _Float16 v8h;
typedef __attribute__((ext_vector_type(8)))  float    v8f;

union Frag { v16h v; v8h h[2]; };

// order-preserving float <-> uint map for atomicMax on signed floats
__device__ __forceinline__ unsigned mapf(float f) {
    unsigned u = __float_as_uint(f);
    return (u & 0x80000000u) ? ~u : (u | 0x80000000u);
}
__device__ __forceinline__ float unmapf(unsigned u) {
    unsigned b = (u & 0x80000000u) ? (u & 0x7FFFFFFFu) : ~u;
    return __uint_as_float(b);
}

// ---------------- K0: transpose+convert W1 -> f16 [N][K]; init segment buffers ----
__global__ void prep_kernel(const float* __restrict__ W1, _Float16* __restrict__ W1T,
                            unsigned* __restrict__ nmax, float* __restrict__ nsum) {
    int idx = blockIdx.x * blockDim.x + threadIdx.x;
    if (idx < K_DIM * N_DIM) {
        int n = idx >> 8;        // / 256
        int k = idx & 255;       // % 256
        W1T[idx] = (_Float16)W1[k * N_DIM + n];
    }
    if (idx < N_NODES) { nmax[idx] = 0u; nsum[idx] = 0.0f; }
}

// ---------------- K1: fused gather + MLP via WMMA + per-node atomic max ----------
// Each wave owns TWO 16-edge M-tiles so every B fragment fetched from W1T feeds
// two v_wmma issues (halves B traffic, doubles wmma per vmem in steady state).
#define WAVES 2                        // waves per block (64 threads)
#define MT 2                           // M-tiles per wave
#define ROWSTRIDE 264                  // 256 + 8 halfs pad (bank stagger)
#define WTILE (16 * ROWSTRIDE)

__global__ void __launch_bounds__(64)
edge_mlp_kernel(const float* __restrict__ h,
                const _Float16* __restrict__ W1T,
                const float* __restrict__ b1,
                const float* __restrict__ W2,
                const float* __restrict__ b2,
                const int* __restrict__ row,
                const int* __restrict__ col,
                float* __restrict__ scores,
                unsigned* __restrict__ nmax) {
    __shared__ _Float16 ldsA[WAVES * MT * WTILE];   // 33792 B
    const int lane = threadIdx.x & 31;
    const int wid  = threadIdx.x >> 5;
    const int wave = blockIdx.x * WAVES + wid;
    const int e0   = wave * (16 * MT);              // first of 32 edges for this wave

    // gather h[row]/h[col] -> LDS f16 tiles [MT][16 rows x 256 K], 2 lanes per edge
    {
        const int e   = lane & 15;
        const int sel = lane >> 4;                  // 0: row half, 1: col half
        #pragma unroll
        for (int t = 0; t < MT; ++t) {
            const int eg  = e0 + t * 16 + e;
            const int node = sel ? col[eg] : row[eg];
            const float4* src = (const float4*)(h + (size_t)node * EMB);
            _Float16* dst = ldsA + (wid * MT + t) * WTILE + e * ROWSTRIDE + sel * EMB;
            #pragma unroll
            for (int i = 0; i < 16; ++i) {
                float4 f0 = src[2 * i], f1 = src[2 * i + 1];
                v8h p;
                p[0] = (_Float16)f0.x; p[1] = (_Float16)f0.y;
                p[2] = (_Float16)f0.z; p[3] = (_Float16)f0.w;
                p[4] = (_Float16)f1.x; p[5] = (_Float16)f1.y;
                p[6] = (_Float16)f1.z; p[7] = (_Float16)f1.w;
                *(v8h*)(dst + i * 8) = p;
            }
        }
    }
    __syncthreads();

    // load A fragments for both tiles (each 16x256 over 8 k-steps) into registers
    const int koff = (lane >> 4) << 3;              // 0 or 8 (halfs)
    Frag a[MT][8];
    {
        const int m = lane & 15;
        #pragma unroll
        for (int t = 0; t < MT; ++t) {
            const _Float16* rp = ldsA + (wid * MT + t) * WTILE + m * ROWSTRIDE;
            #pragma unroll
            for (int kk = 0; kk < 8; ++kk) {
                a[t][kk].h[0] = *(const v8h*)(rp + kk * 32 + koff);
                a[t][kk].h[1] = *(const v8h*)(rp + kk * 32 + 16 + koff);
            }
        }
    }

    float acc[MT][8] = {};
    const int nl = lane & 15;

    for (int nt = 0; nt < 32; ++nt) {
        const int n = nt * 16 + nl;
        const _Float16* bp = W1T + (size_t)n * K_DIM + koff;
        if (nt < 31) __builtin_prefetch(bp + 16 * K_DIM, 0, 3);  // next B tile row (WGP scope)
        v8f c[MT];
        #pragma unroll
        for (int t = 0; t < MT; ++t)
            c[t] = (v8f){0.f, 0.f, 0.f, 0.f, 0.f, 0.f, 0.f, 0.f};
        #pragma unroll
        for (int kk = 0; kk < 8; ++kk) {
            Frag b;
            b.h[0] = *(const v8h*)(bp + kk * 32);
            b.h[1] = *(const v8h*)(bp + kk * 32 + 16);
            #pragma unroll
            for (int t = 0; t < MT; ++t)
                c[t] = __builtin_amdgcn_wmma_f32_16x16x32_f16(false, a[t][kk].v, false, b.v,
                                                              (short)0, c[t], false, false);
        }
        // fused layer 2: relu(hid + b1) * W2, accumulated per lane
        const float bb = b1[n];
        const float ww = W2[n];
        #pragma unroll
        for (int t = 0; t < MT; ++t) {
            #pragma unroll
            for (int v = 0; v < 8; ++v) {
                float x = c[t][v] + bb;
                x = x > 0.f ? x : 0.f;
                acc[t][v] += x * ww;
            }
        }
    }

    // reduce over the 16 lanes of each half (rows v in lanes 0-15, rows v+8 in 16-31)
    #pragma unroll
    for (int t = 0; t < MT; ++t) {
        #pragma unroll
        for (int v = 0; v < 8; ++v) {
            float x = acc[t][v];
            x += __shfl_xor(x, 1, 32);
            x += __shfl_xor(x, 2, 32);
            x += __shfl_xor(x, 4, 32);
            x += __shfl_xor(x, 8, 32);
            acc[t][v] = x;
        }
    }

    if (nl == 0) {                    // lanes 0 and 16 hold the reduced rows
        const int mbase = (lane >> 4) * 8;
        const float bias2 = b2[0];
        #pragma unroll
        for (int t = 0; t < MT; ++t) {
            #pragma unroll
            for (int v = 0; v < 8; ++v) {
                const int e = e0 + t * 16 + mbase + v;
                const float sc = acc[t][v] + bias2;
                scores[e] = sc;
                atomicMax(&nmax[row[e]], mapf(sc));   // segment-max pass, fused
            }
        }
    }
}

// ---------------- K2: segment sum of exp(score - max) -----------------------------
__global__ void seg_sum_kernel(const float* __restrict__ scores,
                               const int* __restrict__ row,
                               const unsigned* __restrict__ nmax,
                               float* __restrict__ nsum) {
    int e = blockIdx.x * blockDim.x + threadIdx.x;
    if (e >= N_EDGES) return;
    int r = row[e];
    float m = unmapf(nmax[r]);
    atomicAdd(&nsum[r], expf(scores[e] - m));
}

// ---------------- K3: softmax -> relaxed bernoulli -> masks -----------------------
__global__ void finalize_kernel(const float* __restrict__ scores,
                                const float* __restrict__ u,
                                const int* __restrict__ row,
                                const int* __restrict__ emask,
                                const int* __restrict__ hier,
                                const unsigned* __restrict__ nmax,
                                const float* __restrict__ nsum,
                                float* __restrict__ out) {
    int e = blockIdx.x * blockDim.x + threadIdx.x;
    if (e >= N_EDGES) return;
    int r = row[e];
    float sc = scores[e];
    float m = unmapf(nmax[r]);
    float s = nsum[r];
    float ex = expf(sc - m);
    float rem = fmaxf(s - ex, 0.0f);
    // logits = log(p) - log1p(-p), p = ex/s  =>  (sc - m) - log(s - ex)
    float logits = (sc - m) - logf(rem);             // rem==0 -> +inf -> y==1
    float lg = logf(u[e]) - log1pf(-u[e]);           // logistic noise
    float z = logits + lg;                            // TEMPERATURE = 1
    float y = 1.0f / (1.0f + expf(-z));
    bool hard = y > 0.5f;                             // RATIO = 0.5
    float yst = hard ? 1.0f : 0.0f;                   // straight-through == y_hard
    int nm = hard ? (hier[0] + 1) : emask[e];
    out[e]               = sc;
    out[N_EDGES + e]     = yst;
    out[2 * N_EDGES + e] = (float)nm;
    out[3 * N_EDGES + e] = (nm > 0) ? sc : 0.0f;      // causal_w
    out[4 * N_EDGES + e] = (nm == -1) ? -sc : 0.0f;   // spu_w
}

extern "C" void kernel_launch(void* const* d_in, const int* in_sizes, int n_in,
                              void* d_out, int out_size, void* d_ws, size_t ws_size,
                              hipStream_t stream) {
    const float* h   = (const float*)d_in[0];
    const float* W1  = (const float*)d_in[1];
    const float* b1  = (const float*)d_in[2];
    const float* W2  = (const float*)d_in[3];
    const float* b2  = (const float*)d_in[4];
    const float* u   = (const float*)d_in[5];
    const int*   row = (const int*)d_in[6];
    const int*   col = (const int*)d_in[7];
    const int*   em  = (const int*)d_in[8];
    const int*   hr  = (const int*)d_in[9];

    char* ws = (char*)d_ws;
    _Float16* W1T  = (_Float16*)ws;                          // 262144 B
    unsigned* nmax = (unsigned*)(ws + 262144);               //  80000 B
    float*    nsum = (float*)(ws + 262144 + 80000);          //  80000 B

    float* out    = (float*)d_out;
    float* scores = out;                                     // first E floats

    prep_kernel<<<(K_DIM * N_DIM + 255) / 256, 256, 0, stream>>>(W1, W1T, nmax, nsum);
    edge_mlp_kernel<<<N_EDGES / (16 * MT * WAVES), 64, 0, stream>>>(
        h, W1T, b1, W2, b2, row, col, scores, nmax);
    seg_sum_kernel<<<(N_EDGES + 255) / 256, 256, 0, stream>>>(scores, row, nmax, nsum);
    finalize_kernel<<<(N_EDGES + 255) / 256, 256, 0, stream>>>(
        scores, u, row, em, hr, nmax, nsum, out);
}
